// WindowAttention_57612691309521
// MI455X (gfx1250) — matrix operand order repaired and checked by reading, not compiled, not run on hardware
//
#include <hip/hip_runtime.h>
#include <hip/hip_bf16.h>

typedef __attribute__((ext_vector_type(16))) __bf16 v16bf;
typedef __attribute__((ext_vector_type(8)))  float  v8f;

#define WHh 7
#define WWw 7
#define N_TOK 49
#define NP 64
#define NHEAD 12
#define CDIM 384
#define HD 32
#define NWIN 64

#define QKV_W_ELTS (3 * CDIM * CDIM)   // 442368
#define PROJ_W_ELTS (CDIM * CDIM)      // 147456
#define O_BYTES ((size_t)2048 * N_TOK * CDIM * 2)        // 77,070,336
#define QKVW_OFF O_BYTES                                  // bf16 qkv_w
#define PROJW_OFF (O_BYTES + (size_t)QKV_W_ELTS * 2)      // bf16 proj_w

// LDS carve-up (single static block, 64512 B < 64 KB):
//   [0      .. 49152)  xs   : 64 x 384 bf16  (x tile, dead after phase 1)
//   [0      .. 17408)  Ss   : 64 x 68  f32   (S scores, overlaid on xs)
//   [17408  .. 26624)  Ps   : 64 x 72  bf16  (softmax probs, overlaid on xs)
//   [49152  .. 54272)  qs   : 64 x 40  bf16
//   [54272  .. 59392)  ks   : 64 x 40  bf16
//   [59392  .. 64512)  vs   : 64 x 40  bf16
#define SMEM_BYTES 64512

static __device__ __forceinline__ v8f wmma_bf16(v16bf a, v16bf b, v8f c) {
  // (neg_a, A, neg_b, B, c_mod, C, reuse_a, reuse_b)
  return __builtin_amdgcn_wmma_f32_16x16x32_bf16(false, a, false, b, (short)0, c,
                                                 false, false);
}

// ---------------------------------------------------------------------------
// One-shot weight conversion f32 -> bf16 (qkv_w then proj_w, concatenated).
// ---------------------------------------------------------------------------
__global__ __launch_bounds__(256)
void cvt_weights_kernel(const float* __restrict__ qkv_w,
                        const float* __restrict__ proj_w,
                        __bf16* __restrict__ qkv_wb,
                        __bf16* __restrict__ proj_wb)
{
  const int i = blockIdx.x * 256 + threadIdx.x;
  if (i < QKV_W_ELTS) qkv_wb[i] = (__bf16)qkv_w[i];
  if (i < PROJ_W_ELTS) proj_wb[i] = (__bf16)proj_w[i];
}

// ---------------------------------------------------------------------------
// Fused QKV-projection + windowed attention. One block per (window, head).
// ---------------------------------------------------------------------------
__global__ __launch_bounds__(256)
void swin_attn_kernel(const float* __restrict__ x,
                      const float* __restrict__ mask,
                      const __bf16* __restrict__ qkv_wb,
                      const float* __restrict__ qkv_b,
                      const float* __restrict__ bias_table,
                      __bf16* __restrict__ o)
{
  __shared__ __align__(16) unsigned char smem[SMEM_BYTES];
  __bf16* xs = (__bf16*)smem;
  float*  Ss = (float*)smem;
  __bf16* Ps = (__bf16*)(smem + 17408);
  __bf16* qs = (__bf16*)(smem + 49152);
  __bf16* ks = (__bf16*)(smem + 54272);
  __bf16* vs = (__bf16*)(smem + 59392);

  const int b    = blockIdx.x / NHEAD;
  const int h    = blockIdx.x % NHEAD;
  const int tid  = threadIdx.x;
  const int wave = tid >> 5;
  const int lane = tid & 31;
  const int r    = lane & 15;   // fragment row/col index
  const int half = lane >> 4;   // K-half selector

  // ---- stage x[b] (49x384 f32) -> LDS bf16, rows 49..63 zero-padded -------
  {
    const float4* xg = (const float4*)(x + (size_t)b * N_TOK * CDIM);
    for (int i = tid; i < NP * (CDIM / 4); i += 256) {
      const int rr = i / (CDIM / 4);
      const int cc = (i % (CDIM / 4)) * 4;
      float4 v = (rr < N_TOK) ? xg[rr * (CDIM / 4) + (cc >> 2)]
                              : make_float4(0.f, 0.f, 0.f, 0.f);
      __bf16* d = xs + rr * CDIM + cc;
      d[0] = (__bf16)v.x; d[1] = (__bf16)v.y;
      d[2] = (__bf16)v.z; d[3] = (__bf16)v.w;
    }
  }
  __syncthreads();

  // ---- phase 1: q/k/v = x @ W^T + b (24 WMMA tiles across 8 waves) --------
  const float scale = 0.17677669529663687f;  // 1/sqrt(32)
  #pragma unroll
  for (int t3 = 0; t3 < 3; ++t3) {
    const int t   = wave + t3 * 8;      // tile id 0..23
    const int m   = t >> 3;             // 0=q 1=k 2=v (wave-uniform)
    const int sub = t & 7;
    const int mt  = sub >> 1;           // M tile 0..3
    const int nt  = sub & 1;            // N tile 0..1
    const int wrow = m * CDIM + h * HD + nt * 16 + r;  // qkv_w row / bias idx
    v8f acc = {};
    for (int ksp = 0; ksp < CDIM / 32; ++ksp) {
      v16bf a, bb;
      const __bf16* xr = xs + (mt * 16 + r) * CDIM + ksp * 32 + half * 16;
      #pragma unroll
      for (int i = 0; i < 8; ++i) a[i] = xr[i];
      #pragma unroll
      for (int i = 0; i < 8; ++i) a[8 + i] = xr[8 + i];
      const __bf16* wr = qkv_wb + (size_t)wrow * CDIM + ksp * 32 + half * 16;
      #pragma unroll
      for (int i = 0; i < 16; ++i) bb[i] = wr[i];
      acc = wmma_bf16(a, bb, acc);
    }
    const float bias = qkv_b[wrow];
    const float sc   = (m == 0) ? scale : 1.0f;
    __bf16* dst = (m == 0) ? qs : (m == 1) ? ks : vs;
    #pragma unroll
    for (int j = 0; j < 8; ++j) {
      const int row = mt * 16 + half * 8 + j;
      dst[row * 40 + nt * 16 + r] = (__bf16)((acc[j] + bias) * sc);
    }
  }
  __syncthreads();

  // ---- phase 2: S = (q*scale) @ k^T  (16 tiles, 2 per wave) ---------------
  #pragma unroll
  for (int t2 = 0; t2 < 2; ++t2) {
    const int tt = wave + t2 * 8;
    const int mt = tt >> 2, nt = tt & 3;
    v16bf a, bb;
    const __bf16* qr = qs + (mt * 16 + r) * 40 + half * 16;
    #pragma unroll
    for (int i = 0; i < 8; ++i) a[i] = qr[i];
    #pragma unroll
    for (int i = 0; i < 8; ++i) a[8 + i] = qr[8 + i];
    const __bf16* kr = ks + (nt * 16 + r) * 40 + half * 16;
    #pragma unroll
    for (int i = 0; i < 16; ++i) bb[i] = kr[i];
    v8f acc = {};
    acc = wmma_bf16(a, bb, acc);
    #pragma unroll
    for (int j = 0; j < 8; ++j)
      Ss[(mt * 16 + half * 8 + j) * 68 + nt * 16 + r] = acc[j];
  }
  __syncthreads();

  // ---- phase 3: +rel-pos bias, +window mask, softmax over 49 cols ---------
  if (tid < NP) {
    const int n = tid;
    __bf16* prow = Ps + n * 72;
    if (n < N_TOK) {
      const int i0 = n / WWw, j0 = n % WWw;
      const float* mrow =
          mask + (size_t)(b & (NWIN - 1)) * N_TOK * N_TOK + n * N_TOK;
      float* srow = Ss + n * 68;
      float mx = -3.0e38f;
      for (int mc = 0; mc < N_TOK; ++mc) {
        const int i1 = mc / WWw, j1 = mc % WWw;
        const int rel = (i0 - i1 + WHh - 1) * (2 * WWw - 1) + (j0 - j1 + WWw - 1);
        const float s = srow[mc] + bias_table[rel * NHEAD + h] + mrow[mc];
        srow[mc] = s;
        mx = fmaxf(mx, s);
      }
      float sum = 0.f;
      for (int mc = 0; mc < N_TOK; ++mc) {
        const float e = __expf(srow[mc] - mx);
        srow[mc] = e;
        sum += e;
      }
      const float rinv = 1.f / sum;
      for (int mc = 0; mc < N_TOK; ++mc) prow[mc] = (__bf16)(srow[mc] * rinv);
      for (int mc = N_TOK; mc < NP; ++mc) prow[mc] = (__bf16)0.f;
    } else {
      for (int mc = 0; mc < NP; ++mc) prow[mc] = (__bf16)0.f;
    }
  }
  __syncthreads();

  // ---- phase 4: O = P @ V  (8 tiles, 1 per wave, K=64 in 2 steps) ---------
  {
    const int mt = wave >> 1, nt = wave & 1;
    v8f acc = {};
    #pragma unroll
    for (int ksp = 0; ksp < 2; ++ksp) {
      v16bf a, bb;
      const __bf16* pr = Ps + (mt * 16 + r) * 72 + ksp * 32 + half * 16;
      #pragma unroll
      for (int i = 0; i < 8; ++i) a[i] = pr[i];
      #pragma unroll
      for (int i = 0; i < 8; ++i) a[8 + i] = pr[8 + i];
      #pragma unroll
      for (int i = 0; i < 16; ++i)
        bb[i] = vs[(ksp * 32 + half * 16 + i) * 40 + nt * 16 + r];
      acc = wmma_bf16(a, bb, acc);
    }
    #pragma unroll
    for (int j = 0; j < 8; ++j) {
      const int row = mt * 16 + half * 8 + j;
      if (row < N_TOK)
        o[((size_t)b * N_TOK + row) * CDIM + h * HD + nt * 16 + r] =
            (__bf16)acc[j];
    }
  }
}

// ---------------------------------------------------------------------------
// Output projection: out = o(bf16) @ proj_w^T + proj_b.  One 16x32 tile/wave.
// M = 2048*49 = 100352 = 6272 tiles; N groups = 384/32 = 12; 75264 wave-tasks.
// ---------------------------------------------------------------------------
__global__ __launch_bounds__(256)
void swin_proj_kernel(const __bf16* __restrict__ o,
                      const __bf16* __restrict__ proj_wb,
                      const float* __restrict__ proj_b,
                      float* __restrict__ out)
{
  const int tid  = threadIdx.x;
  const int wave = tid >> 5;
  const int lane = tid & 31;
  const int r    = lane & 15;
  const int half = lane >> 4;
  const int wid   = blockIdx.x * 8 + wave;
  const int mtile = wid / 12;
  const int ng    = wid % 12;

  v8f acc0 = {}, acc1 = {};
  for (int ksp = 0; ksp < CDIM / 32; ++ksp) {
    v16bf a, b0, b1;
    const __bf16* ar =
        o + ((size_t)mtile * 16 + r) * CDIM + ksp * 32 + half * 16;
    #pragma unroll
    for (int i = 0; i < 8; ++i) a[i] = ar[i];
    #pragma unroll
    for (int i = 0; i < 8; ++i) a[8 + i] = ar[8 + i];
    const __bf16* w0 =
        proj_wb + (size_t)(ng * 32 + r) * CDIM + ksp * 32 + half * 16;
    const __bf16* w1 = w0 + (size_t)16 * CDIM;
    #pragma unroll
    for (int i = 0; i < 16; ++i) b0[i] = w0[i];
    #pragma unroll
    for (int i = 0; i < 16; ++i) b1[i] = w1[i];
    acc0 = wmma_bf16(a, b0, acc0);
    acc1 = wmma_bf16(a, b1, acc1);
  }
  #pragma unroll
  for (int j = 0; j < 8; ++j) {
    const int row = mtile * 16 + half * 8 + j;
    const int c0  = ng * 32 + r;
    out[(size_t)row * CDIM + c0]      = acc0[j] + proj_b[c0];
    out[(size_t)row * CDIM + c0 + 16] = acc1[j] + proj_b[c0 + 16];
  }
}

extern "C" void kernel_launch(void* const* d_in, const int* in_sizes, int n_in,
                              void* d_out, int out_size, void* d_ws, size_t ws_size,
                              hipStream_t stream) {
  (void)in_sizes; (void)n_in; (void)out_size; (void)ws_size;
  const float* x          = (const float*)d_in[0];
  const float* mask       = (const float*)d_in[1];
  const float* qkv_w      = (const float*)d_in[2];
  const float* qkv_b      = (const float*)d_in[3];
  const float* proj_w     = (const float*)d_in[4];
  const float* proj_b     = (const float*)d_in[5];
  const float* bias_table = (const float*)d_in[6];

  unsigned char* ws = (unsigned char*)d_ws;
  __bf16* o       = (__bf16*)ws;                 // 77,070,336 B
  __bf16* qkv_wb  = (__bf16*)(ws + QKVW_OFF);    //    884,736 B
  __bf16* proj_wb = (__bf16*)(ws + PROJW_OFF);   //    294,912 B

  // one-shot f32 -> bf16 weight conversion
  cvt_weights_kernel<<<(QKV_W_ELTS + 255) / 256, 256, 0, stream>>>(
      qkv_w, proj_w, qkv_wb, proj_wb);

  swin_attn_kernel<<<2048 * NHEAD, 256, 0, stream>>>(x, mask, qkv_wb, qkv_b,
                                                     bias_table, o);
  // 6272 M-tiles * 12 N-groups = 75264 wave tasks / 8 waves per block
  swin_proj_kernel<<<75264 / 8, 256, 0, stream>>>(o, proj_wb, proj_b,
                                                  (float*)d_out);
}
